// BlockDiffusionDecoder_18141941858992
// MI455X (gfx1250) — matrix operand order repaired and checked by reading, not compiled, not run on hardware
//
#include <hip/hip_runtime.h>
#include <math.h>

typedef _Float16 f16;
typedef __attribute__((ext_vector_type(16))) _Float16 v16h;
typedef __attribute__((ext_vector_type(8)))  float    v8f;

// ---------------- model constants ----------------
static constexpr int Bq   = 2;
static constexpr int Dm   = 1024;
static constexpr int Hh   = 16;
static constexpr int NLY  = 6;
static constexpr int DFFc = 4096;
static constexpr int Vv   = 50000;
static constexpr int Vp   = Vv + 1;      // 50001
static constexpr int HDc  = 64;          // Dm / Hh
static constexpr int BLKc = 4;
static constexpr int Lc   = 256;         // L_RAW + 1
static constexpr int T2c  = 512;         // 2*L
static constexpr int SEnc = 300;
static constexpr int M1   = Bq * T2c;    // 1024 rows of main stream
static constexpr long DD  = (long)Dm * Dm;

// ---------------- block reductions ----------------
__device__ __forceinline__ float blk_reduce_add(float v, float* red, int tid) {
  red[tid] = v; __syncthreads();
  for (int o = 128; o > 0; o >>= 1) { if (tid < o) red[tid] += red[tid + o]; __syncthreads(); }
  float r = red[0]; __syncthreads();
  return r;
}
__device__ __forceinline__ float blk_reduce_max(float v, float* red, int tid) {
  red[tid] = v; __syncthreads();
  for (int o = 128; o > 0; o >>= 1) { if (tid < o) red[tid] = fmaxf(red[tid], red[tid + o]); __syncthreads(); }
  float r = red[0]; __syncthreads();
  return r;
}

// Assemble a 16-half fragment from two 16-byte LDS runs (emits 2x ds_load_b128).
__device__ __forceinline__ v16h load_frag2(const f16* p0, const f16* p1) {
  union { v16h v; uint4 q[2]; } u;
  u.q[0] = *(const uint4*)p0;
  u.q[1] = *(const uint4*)p1;
  return u.v;
}

// ---------------- generic batched f16 WMMA GEMM ----------------
// C[z] = A[z] * op(B[z]) (+bias[n]) (+res[z][m,n]) , fp32 accumulate.
// NT=false: B is K x N row-major (ldb); NT=true: B is N x K row-major (ldb).
// GUARD=false requires M%128==0, N%64==0, K%32==0 (all big aligned GEMMs).
// Block: 256 threads = 8 wave32; tile 128(M) x 64(N), K-step 32.
// Each wave owns a 32x32 sub-tile: 2 A-frags x 2 B-frags -> 4 WMMA per K-step.
template <bool NT, bool GUARD>
__global__ __launch_bounds__(256) void gemm_f16_wmma(
    const f16* __restrict__ A, const f16* __restrict__ Bm,
    const float* __restrict__ bias, const float* __restrict__ res,
    float* __restrict__ C,
    int M, int N, int K, int lda, int ldb, int ldc,
    long sA, long sB, long sC)
{
  constexpr int LDA = 40;   // halves per As row (32 + pad) -> 80 B, 16B-aligned rows
  constexpr int LDB = 40;   // halves per Bs row (n-major: Bs[n][k])
  __shared__ __attribute__((aligned(16))) f16 As[128 * LDA];  // 10240 B
  __shared__ __attribute__((aligned(16))) f16 Bs[64 * LDB];   //  5120 B

  const int tid  = threadIdx.x;
  const int lane = tid & 31;
  const int wave = tid >> 5;
  const int z    = blockIdx.z;
  const int m0   = blockIdx.y * 128;
  const int n0   = blockIdx.x * 64;

  const f16*   Ab = A + (long)z * sA;
  const f16*   Bb = Bm + (long)z * sB;
  float*       Cb = C + (long)z * sC;
  const float* Rb = res ? (res + (long)z * sC) : nullptr;

  const int mw = (wave & 3) * 32;   // wave's 32-row slice
  const int nw = (wave >> 2) * 32;  // wave's 32-col slice

  v8f acc00 = {}, acc01 = {}, acc10 = {}, acc11 = {};

  // A staging: each thread owns one row, 16 contiguous halves (two uint4)
  const int arow = tid >> 1;
  const int acb  = (tid & 1) * 16;
  // B staging coords
  const int ntn = tid >> 2, ntk = (tid & 3) * 8;   // NT: 8 contiguous k at fixed n
  const int nnn = tid & 63, nnk = (tid >> 6) * 8;  // NN: 8 k-strided at fixed n (lane-coalesced)

  for (int k0 = 0; k0 < K; k0 += 32) {
    __syncthreads();
    // ---- stage A tile 128x32 ----
    {
      const long gm = m0 + arow;
      if (!GUARD || (gm < M && (k0 + acb + 15) < K)) {
        *(uint4*)&As[arow * LDA + acb]     = *(const uint4*)&Ab[gm * (long)lda + k0 + acb];
        *(uint4*)&As[arow * LDA + acb + 8] = *(const uint4*)&Ab[gm * (long)lda + k0 + acb + 8];
        if (k0 + 32 < K) __builtin_prefetch(&Ab[gm * (long)lda + k0 + 32 + acb], 0, 0);
      } else {
#pragma unroll
        for (int u = 0; u < 16; ++u) {
          f16 v = (f16)0.f;
          if (gm < M && (k0 + acb + u) < K) v = Ab[gm * (long)lda + k0 + acb + u];
          As[arow * LDA + acb + u] = v;
        }
      }
    }
    // ---- stage B tile into n-major LDS: Bs[n][k] ----
    if (NT) {
      const long gn = n0 + ntn;
      if (!GUARD || (gn < N && (k0 + ntk + 7) < K)) {
        *(uint4*)&Bs[ntn * LDB + ntk] = *(const uint4*)&Bb[gn * (long)ldb + k0 + ntk];
      } else {
#pragma unroll
        for (int u = 0; u < 8; ++u) {
          f16 v = (f16)0.f;
          if (gn < N && (k0 + ntk + u) < K) v = Bb[gn * (long)ldb + k0 + ntk + u];
          Bs[ntn * LDB + ntk + u] = v;
        }
      }
    } else {
      union { uint4 q; f16 v[8]; } tb;
#pragma unroll
      for (int u = 0; u < 8; ++u) {
        f16 v = (f16)0.f;
        const long gk = k0 + nnk + u;
        if (!GUARD || (gk < K && (n0 + nnn) < N)) v = Bb[gk * (long)ldb + n0 + nnn];
        tb.v[u] = v;
      }
      *(uint4*)&Bs[nnn * LDB + nnk] = tb.q;  // aligned 16B store
    }
    __syncthreads();

    // ---- fragments (ISA 16-bit layouts) ----
    // A 16x32: lane m = lane&15; lanes 0-15 take K {0..7,16..23}, lanes 16-31 take +8.
    const int ml  = lane & 15;
    const int kh2 = (lane >> 4) * 8;
    const f16* Ar0 = &As[(mw + ml) * LDA + kh2];
    const f16* Ar1 = &As[(mw + 16 + ml) * LDA + kh2];
    v16h a0 = load_frag2(Ar0, Ar0 + 16);
    v16h a1 = load_frag2(Ar1, Ar1 + 16);
    // B 32x16: lane n = lane&15; lanes 0-15 take K {0..7,16..23}, lanes 16-31 take +8.
    const int khb = (lane >> 4) * 8;
    const f16* Br0 = &Bs[(nw + ml) * LDB + khb];
    const f16* Br1 = &Bs[(nw + 16 + ml) * LDB + khb];
    v16h b0 = load_frag2(Br0, Br0 + 16);
    v16h b1 = load_frag2(Br1, Br1 + 16);

    acc00 = __builtin_amdgcn_wmma_f32_16x16x32_f16(false, a0, false, b0, (short)0, acc00, false, false);
    acc01 = __builtin_amdgcn_wmma_f32_16x16x32_f16(false, a0, false, b1, (short)0, acc01, false, false);
    acc10 = __builtin_amdgcn_wmma_f32_16x16x32_f16(false, a1, false, b0, (short)0, acc10, false, false);
    acc11 = __builtin_amdgcn_wmma_f32_16x16x32_f16(false, a1, false, b1, (short)0, acc11, false, false);
  }

  // ---- store (f32 C/D layout: lane n = lane&15, VGPR j -> m = j + (lane>>4)*8) ----
  const int ml   = lane & 15;
  const int mhi  = (lane >> 4) * 8;
#pragma unroll
  for (int mi = 0; mi < 2; ++mi) {
    const v8f* accn0 = mi == 0 ? &acc00 : &acc10;
    const v8f* accn1 = mi == 0 ? &acc01 : &acc11;
    const int crow = m0 + mw + mi * 16 + mhi;
#pragma unroll
    for (int j = 0; j < 8; ++j) {
      const int gm = crow + j;
      if (!GUARD || gm < M) {
        const int gn0 = n0 + nw + ml;
        if (!GUARD || gn0 < N) {
          float v = (*accn0)[j];
          if (bias) v += bias[gn0];
          if (Rb)   v += Rb[(long)gm * ldc + gn0];
          Cb[(long)gm * ldc + gn0] = v;
        }
        const int gn1 = n0 + nw + ml + 16;
        if (!GUARD || gn1 < N) {
          float v = (*accn1)[j];
          if (bias) v += bias[gn1];
          if (Rb)   v += Rb[(long)gm * ldc + gn1];
          Cb[(long)gm * ldc + gn1] = v;
        }
      }
    }
  }
}

// ---------------- prep: x0, t, masked, xt, ids ----------------
__global__ void prep_kernel(const float* __restrict__ t_rand, const float* __restrict__ mask_rand,
                            const int* __restrict__ labels,
                            int* __restrict__ ids, int* __restrict__ x0v,
                            float* __restrict__ tt, float* __restrict__ mkf)
{
  int i = blockIdx.x * blockDim.x + threadIdx.x;
  if (i >= Bq * Lc) return;
  const int b = i / Lc, p = i % Lc;
  const int x0 = (p == 0) ? 0 : labels[b * (Lc - 1) + (p - 1)];
  const bool tmask = (x0 != 1) && (p != 0);   // PAD==1, BOS pos forced false
  const int nb = Lc / BLKc;                    // 64
  const int Bn = Bq * nb;                      // 128
  const int j  = p / BLKc;
  float t = t_rand[b * nb + j] / (float)Bn + (float)(b * nb + j) / (float)Bn;
  t = t - floorf(t);
  t = t * (1.0f - 0.001f) + 0.001f;
  const bool masked = (mask_rand[b * Lc + p] < t) && tmask;
  const int xt = masked ? Vv : x0;
  ids[b * T2c + p]      = xt;
  ids[b * T2c + Lc + p] = x0;
  x0v[i] = x0;
  tt[i]  = t;
  mkf[i] = masked ? 1.f : 0.f;
}

// ---------------- embedding ----------------
__global__ void embed_kernel(const int* __restrict__ ids, const float* __restrict__ emb,
                             const float* __restrict__ pos, float* __restrict__ out)
{
  long i = (long)blockIdx.x * blockDim.x + threadIdx.x;
  const long total = (long)Bq * T2c * Dm;
  if (i >= total) return;
  const long row = i / Dm;
  const int  d   = (int)(i % Dm);
  const int  p   = (int)(row % T2c);
  const int  pid = (p < Lc ? p : p - Lc) + 2;
  out[i] = emb[(long)ids[row] * Dm + d] * 32.0f + pos[(long)pid * Dm + d];  // sqrt(1024)=32
}

// ---------------- LayerNorm over D=1024, one block per row ----------------
__global__ __launch_bounds__(256) void ln_kernel(const float* __restrict__ x,
                                                 const float* __restrict__ sc,
                                                 const float* __restrict__ bi,
                                                 float* __restrict__ outf,
                                                 f16* __restrict__ outh)
{
  __shared__ float red[256];
  const int row = blockIdx.x, tid = threadIdx.x;
  const float* xr = x + (long)row * Dm;
  float loc[4]; float s = 0.f;
#pragma unroll
  for (int u = 0; u < 4; ++u) { loc[u] = xr[tid + u * 256]; s += loc[u]; }
  const float mean = blk_reduce_add(s, red, tid) * (1.0f / Dm);
  float vs = 0.f;
#pragma unroll
  for (int u = 0; u < 4; ++u) { const float d = loc[u] - mean; vs += d * d; }
  const float var  = blk_reduce_add(vs, red, tid) * (1.0f / Dm);
  const float rstd = rsqrtf(var + 1e-5f);
#pragma unroll
  for (int u = 0; u < 4; ++u) {
    const int c = tid + u * 256;
    const float y = (loc[u] - mean) * rstd * sc[c] + bi[c];
    if (outf) outf[(long)row * Dm + c] = y;
    if (outh) outh[(long)row * Dm + c] = (f16)y;
  }
}

// ---------------- elementwise helpers ----------------
__global__ void cvt_kernel(const float* __restrict__ in, f16* __restrict__ out, long n) {
  long i = (long)blockIdx.x * blockDim.x + threadIdx.x;
  if (i < n) out[i] = (f16)in[i];
}
__global__ void gelu_kernel(const float* __restrict__ in, f16* __restrict__ out, long n) {
  long i = (long)blockIdx.x * blockDim.x + threadIdx.x;
  if (i < n) { const float x = in[i]; out[i] = (f16)(0.5f * x * (1.f + erff(x * 0.70710678118f))); }
}
// [Bq*T, Dm] fp32 -> [Bq*Hh, T, HD] f16 (with optional scale, e.g. HD^-0.5 for q)
__global__ void split_heads(const float* __restrict__ in, f16* __restrict__ out, int T, float scale) {
  long i = (long)blockIdx.x * blockDim.x + threadIdx.x;
  const long total = (long)Bq * T * Dm;
  if (i >= total) return;
  const long row = i / Dm; const int d = (int)(i % Dm);
  const int b = (int)(row / T), t = (int)(row % T);
  const int hh = d / HDc, dd = d % HDc;
  out[(((long)(b * Hh + hh)) * T + t) * HDc + dd] = (f16)(in[i] * scale);
}
// [Bq*Hh, T2, HD] fp32 -> [Bq*T2, Dm] f16
__global__ void merge_heads(const float* __restrict__ in, f16* __restrict__ out) {
  long i = (long)blockIdx.x * blockDim.x + threadIdx.x;
  const long total = (long)Bq * T2c * Dm;
  if (i >= total) return;
  const long row = i / Dm; const int d = (int)(i % Dm);
  const int b = (int)(row / T2c), t = (int)(row % T2c);
  const int hh = d / HDc, dd = d % HDc;
  out[i] = (f16)in[(((long)(b * Hh + hh)) * T2c + t) * HDc + dd];
}

// ---------------- softmax with analytic BD3LM mask ----------------
__global__ __launch_bounds__(256) void softmax_self(const float* __restrict__ s, f16* __restrict__ a) {
  __shared__ float vals[T2c];
  __shared__ float red[256];
  const int row = blockIdx.x;            // Bq*Hh*T2 rows
  const int qi  = row % T2c;
  const int tid = threadIdx.x;
  const float* sr = s + (long)row * T2c;
  const bool fq  = qi >= Lc;
  const int  bqi = (fq ? qi - Lc : qi) >> 2;
  float mx = -3.4e38f;
  for (int c = tid; c < T2c; c += 256) {
    const bool fk = c >= Lc;
    const int  bk = (fk ? c - Lc : c) >> 2;
    const bool can = ((bqi == bk) && (fq == fk)) || ((bqi > bk) && fk && !fq) || ((bqi >= bk) && fk && fq);
    const float v = can ? sr[c] : -3.4e38f;
    vals[c] = v;
    mx = fmaxf(mx, v);
  }
  mx = blk_reduce_max(mx, red, tid);
  float sum = 0.f;
  for (int c = tid; c < T2c; c += 256) { const float e = expf(vals[c] - mx); vals[c] = e; sum += e; }
  sum = blk_reduce_add(sum, red, tid);
  const float inv = 1.f / sum;
  for (int c = tid; c < T2c; c += 256) a[(long)row * T2c + c] = (f16)(vals[c] * inv);
}

// ---------------- softmax with encoder padding mask (300 cols, ld 512) ----------------
__global__ __launch_bounds__(256) void softmax_cross(const float* __restrict__ s,
                                                     const int* __restrict__ enc_mask,
                                                     f16* __restrict__ a) {
  __shared__ float vals[T2c];
  __shared__ float red[256];
  const int row = blockIdx.x;
  const int b   = row / (Hh * T2c);
  const int tid = threadIdx.x;
  const float* sr = s + (long)row * T2c;
  float mx = -3.4e38f;
  for (int c = tid; c < SEnc; c += 256) {
    const float v = (enc_mask[b * SEnc + c] == 1) ? sr[c] : -3.4e38f;
    vals[c] = v;
    mx = fmaxf(mx, v);
  }
  mx = blk_reduce_max(mx, red, tid);
  float sum = 0.f;
  for (int c = tid; c < SEnc; c += 256) { const float e = expf(vals[c] - mx); vals[c] = e; sum += e; }
  sum = blk_reduce_add(sum, red, tid);
  const float inv = 1.f / fmaxf(sum, 1e-30f);
  for (int c = tid; c < SEnc; c += 256) a[(long)row * T2c + c] = (f16)(vals[c] * inv);
}

// ---------------- loss ----------------
__global__ void zero_acc(float* acc) { if (threadIdx.x == 0 && blockIdx.x == 0) { acc[0] = 0.f; acc[1] = 0.f; } }

__global__ __launch_bounds__(256) void loss_kernel(const float* __restrict__ logits,
                                                   const int* __restrict__ x0v,
                                                   const float* __restrict__ tt,
                                                   const float* __restrict__ mkf,
                                                   float* __restrict__ acc) {
  __shared__ float red[256];
  const int r   = blockIdx.x;          // Bq*Lc rows
  const int tid = threadIdx.x;
  const float* lr = logits + (long)r * Vp;
  float mx = -3.4e38f;
  for (int c = tid; c < Vp; c += 256) mx = fmaxf(mx, lr[c]);
  mx = blk_reduce_max(mx, red, tid);
  float sum = 0.f;
  for (int c = tid; c < Vp; c += 256) sum += expf(lr[c] - mx);
  sum = blk_reduce_add(sum, red, tid);
  if (tid == 0) {
    const float lse = mx + logf(sum);
    const int   tgt = x0v[r];
    const float nll = lse - lr[tgt];
    const float w   = 1.f / fmaxf(tt[r], 1e-6f);
    const float lm  = mkf[r];
    atomicAdd(&acc[0], nll * w * lm);
    atomicAdd(&acc[1], lm);
  }
}
__global__ void final_kernel(const float* __restrict__ acc, float* __restrict__ out) {
  if (threadIdx.x == 0 && blockIdx.x == 0) out[0] = acc[0] / fmaxf(acc[1], 1.0f);
}

// ---------------- host-side launch helpers ----------------
static inline void launch_gemm(hipStream_t st, const f16* A, const f16* Bm,
                               const float* bias, const float* res, float* C,
                               int M, int N, int K, int lda, int ldb, int ldc,
                               long sA, long sB, long sC, int batch, bool nt) {
  dim3 g((N + 63) / 64, (M + 127) / 128, batch);
  const bool aligned = (M % 128 == 0) && (N % 64 == 0) && (K % 32 == 0);
  if (nt) {
    if (aligned) gemm_f16_wmma<true , false><<<g, 256, 0, st>>>(A, Bm, bias, res, C, M, N, K, lda, ldb, ldc, sA, sB, sC);
    else         gemm_f16_wmma<true , true ><<<g, 256, 0, st>>>(A, Bm, bias, res, C, M, N, K, lda, ldb, ldc, sA, sB, sC);
  } else {
    if (aligned) gemm_f16_wmma<false, false><<<g, 256, 0, st>>>(A, Bm, bias, res, C, M, N, K, lda, ldb, ldc, sA, sB, sC);
    else         gemm_f16_wmma<false, true ><<<g, 256, 0, st>>>(A, Bm, bias, res, C, M, N, K, lda, ldb, ldc, sA, sB, sC);
  }
}
static inline int gblk(long n) { return (int)((n + 255) / 256); }

extern "C" void kernel_launch(void* const* d_in, const int* in_sizes, int n_in,
                              void* d_out, int out_size, void* d_ws, size_t ws_size,
                              hipStream_t stream) {
  (void)in_sizes; (void)n_in; (void)out_size; (void)ws_size;
  // ---- inputs (setup_inputs order) ----
  const float* enc_hidden = (const float*)d_in[0];
  const float* t_rand     = (const float*)d_in[1];
  const float* mask_rand  = (const float*)d_in[2];
  const float* embed_tok  = (const float*)d_in[3];
  const float* pos_embed  = (const float*)d_in[4];
  const float* ln_emb_s   = (const float*)d_in[5];
  const float* ln_emb_b   = (const float*)d_in[6];
  const float* Wqkv       = (const float*)d_in[7];
  const float* bqkv       = (const float*)d_in[8];
  const float* Wo         = (const float*)d_in[9];
  const float* bo         = (const float*)d_in[10];
  const float* ln1_s      = (const float*)d_in[11];
  const float* ln1_b      = (const float*)d_in[12];
  const float* Wcq        = (const float*)d_in[13];
  const float* bcq        = (const float*)d_in[14];
  const float* Wckv       = (const float*)d_in[15];
  const float* bckv       = (const float*)d_in[16];
  const float* Wco        = (const float*)d_in[17];
  const float* bco        = (const float*)d_in[18];
  const float* ln2_s      = (const float*)d_in[19];
  const float* ln2_b      = (const float*)d_in[20];
  const float* fc1        = (const float*)d_in[21];
  const float* b1         = (const float*)d_in[22];
  const float* fc2        = (const float*)d_in[23];
  const float* b2         = (const float*)d_in[24];
  const float* ln3_s      = (const float*)d_in[25];
  const float* ln3_b      = (const float*)d_in[26];
  const float* lnf_s      = (const float*)d_in[27];
  const float* lnf_b      = (const float*)d_in[28];
  const float* lm_head    = (const float*)d_in[29];
  const int*   labels     = (const int*)d_in[30];
  const int*   enc_mask   = (const int*)d_in[31];

  // ---- workspace carve-up ----
  char* wp = (char*)d_ws;
  auto alloc = [&](size_t bytes) -> char* {
    char* p = wp; wp += (bytes + 255) & ~(size_t)255; return p;
  };
  float* h     = (float*)alloc((size_t)M1 * Dm * 4);
  float* tmp   = (float*)alloc((size_t)M1 * DFFc * 4);
  f16*   hn    = (f16*)  alloc((size_t)M1 * Dm * 2);
  f16*   attnb = (f16*)  alloc((size_t)M1 * Dm * 2);
  f16*   ge    = (f16*)  alloc((size_t)M1 * DFFc * 2);
  f16*   qh    = (f16*)  alloc((size_t)Bq * Hh * T2c * HDc * 2);
  f16*   kh    = (f16*)  alloc((size_t)Bq * Hh * T2c * HDc * 2);
  f16*   vh    = (f16*)  alloc((size_t)Bq * Hh * T2c * HDc * 2);
  float* sbuf  = (float*)alloc((size_t)Bq * Hh * T2c * T2c * 4);
  f16*   abuf  = (f16*)  alloc((size_t)Bq * Hh * T2c * T2c * 2);
  float* oh    = (float*)alloc((size_t)Bq * Hh * T2c * HDc * 4);
  f16*   ench  = (f16*)  alloc((size_t)Bq * SEnc * Dm * 2);
  int*   ids   = (int*)  alloc((size_t)Bq * T2c * 4);
  int*   x0v   = (int*)  alloc((size_t)Bq * Lc * 4);
  float* tt    = (float*)alloc((size_t)Bq * Lc * 4);
  float* mkf   = (float*)alloc((size_t)Bq * Lc * 4);
  float* accp  = (float*)alloc(256);
  float* logits= (float*)alloc((size_t)Bq * Lc * Vp * 4);
  // f16 weight arena
  f16* wQKV = (f16*)alloc((size_t)NLY * 3 * DD * 2);
  f16* wO   = (f16*)alloc((size_t)NLY * DD * 2);
  f16* wCQ  = (f16*)alloc((size_t)NLY * DD * 2);
  f16* wCKV = (f16*)alloc((size_t)NLY * 2 * DD * 2);
  f16* wCO  = (f16*)alloc((size_t)NLY * DD * 2);
  f16* wF1  = (f16*)alloc((size_t)NLY * Dm * DFFc * 2);
  f16* wF2  = (f16*)alloc((size_t)NLY * Dm * DFFc * 2);
  f16* wLM  = (f16*)alloc((size_t)Vp * Dm * 2);

  auto cvt = [&](const float* src, f16* dst, long n) {
    cvt_kernel<<<gblk(n), 256, 0, stream>>>(src, dst, n);
  };
  // ---- weight / encoder conversion (fp32 -> f16, once per launch) ----
  cvt(Wqkv,    wQKV, (long)NLY * 3 * DD);
  cvt(Wo,      wO,   (long)NLY * DD);
  cvt(Wcq,     wCQ,  (long)NLY * DD);
  cvt(Wckv,    wCKV, (long)NLY * 2 * DD);
  cvt(Wco,     wCO,  (long)NLY * DD);
  cvt(fc1,     wF1,  (long)NLY * Dm * DFFc);
  cvt(fc2,     wF2,  (long)NLY * Dm * DFFc);
  cvt(lm_head, wLM,  (long)Vp * Dm);
  cvt(enc_hidden, ench, (long)Bq * SEnc * Dm);

  // ---- prep + embedding + embed-LN ----
  prep_kernel<<<gblk(Bq * Lc), 256, 0, stream>>>(t_rand, mask_rand, labels, ids, x0v, tt, mkf);
  embed_kernel<<<gblk((long)M1 * Dm), 256, 0, stream>>>(ids, embed_tok, pos_embed, tmp);
  ln_kernel<<<M1, 256, 0, stream>>>(tmp, ln_emb_s, ln_emb_b, h, nullptr);

  const float qscale = 0.125f;  // HD^-0.5
  const long  sQ  = (long)T2c * HDc;    // per-(b,h) q/k/v stride (self)
  const long  sKc = (long)SEnc * HDc;   // cross kv stride
  const long  sS  = (long)T2c * T2c;    // score/prob stride
  const int   ZB  = Bq * Hh;            // 32 batched heads

  for (int i = 0; i < NLY; ++i) {
    // ================= self attention =================
    ln_kernel<<<M1, 256, 0, stream>>>(h, ln1_s + (long)i * Dm, ln1_b + (long)i * Dm, nullptr, hn);
    launch_gemm(stream, hn, wQKV + (long)(i * 3 + 0) * DD, bqkv + (long)(i * 3 + 0) * Dm, nullptr, tmp,
                M1, Dm, Dm, Dm, Dm, Dm, 0, 0, 0, 1, false);
    split_heads<<<gblk((long)M1 * Dm), 256, 0, stream>>>(tmp, qh, T2c, qscale);
    launch_gemm(stream, hn, wQKV + (long)(i * 3 + 1) * DD, bqkv + (long)(i * 3 + 1) * Dm, nullptr, tmp,
                M1, Dm, Dm, Dm, Dm, Dm, 0, 0, 0, 1, false);
    split_heads<<<gblk((long)M1 * Dm), 256, 0, stream>>>(tmp, kh, T2c, 1.0f);
    launch_gemm(stream, hn, wQKV + (long)(i * 3 + 2) * DD, bqkv + (long)(i * 3 + 2) * Dm, nullptr, tmp,
                M1, Dm, Dm, Dm, Dm, Dm, 0, 0, 0, 1, false);
    split_heads<<<gblk((long)M1 * Dm), 256, 0, stream>>>(tmp, vh, T2c, 1.0f);
    // scores: s = q @ k^T   (batched over 32 heads, NT)  [512x512x64, aligned]
    launch_gemm(stream, qh, kh, nullptr, nullptr, sbuf,
                T2c, T2c, HDc, HDc, HDc, T2c, sQ, sQ, sS, ZB, true);
    softmax_self<<<ZB * T2c, 256, 0, stream>>>(sbuf, abuf);
    // o = a @ v (batched NN) [512x64x512, aligned]
    launch_gemm(stream, abuf, vh, nullptr, nullptr, oh,
                T2c, HDc, T2c, T2c, HDc, HDc, sS, sQ, sQ, ZB, false);
    merge_heads<<<gblk((long)M1 * Dm), 256, 0, stream>>>(oh, attnb);
    // out-proj + bias + residual into h
    launch_gemm(stream, attnb, wO + (long)i * DD, bo + (long)i * Dm, h, h,
                M1, Dm, Dm, Dm, Dm, Dm, 0, 0, 0, 1, false);

    // ================= cross attention =================
    ln_kernel<<<M1, 256, 0, stream>>>(h, ln2_s + (long)i * Dm, ln2_b + (long)i * Dm, nullptr, hn);
    launch_gemm(stream, hn, wCQ + (long)i * DD, bcq + (long)i * Dm, nullptr, tmp,
                M1, Dm, Dm, Dm, Dm, Dm, 0, 0, 0, 1, false);
    split_heads<<<gblk((long)M1 * Dm), 256, 0, stream>>>(tmp, qh, T2c, qscale);
    launch_gemm(stream, ench, wCKV + (long)(i * 2 + 0) * DD, bckv + (long)(i * 2 + 0) * Dm, nullptr, tmp,
                Bq * SEnc, Dm, Dm, Dm, Dm, Dm, 0, 0, 0, 1, false);   // M=600 -> guarded
    split_heads<<<gblk((long)Bq * SEnc * Dm), 256, 0, stream>>>(tmp, kh, SEnc, 1.0f);
    launch_gemm(stream, ench, wCKV + (long)(i * 2 + 1) * DD, bckv + (long)(i * 2 + 1) * Dm, nullptr, tmp,
                Bq * SEnc, Dm, Dm, Dm, Dm, Dm, 0, 0, 0, 1, false);   // M=600 -> guarded
    split_heads<<<gblk((long)Bq * SEnc * Dm), 256, 0, stream>>>(tmp, vh, SEnc, 1.0f);
    // scores over 300 enc tokens (N=300 -> guarded; ldc padded to 512)
    launch_gemm(stream, qh, kh, nullptr, nullptr, sbuf,
                T2c, SEnc, HDc, HDc, HDc, T2c, sQ, sKc, sS, ZB, true);
    softmax_cross<<<ZB * T2c, 256, 0, stream>>>(sbuf, enc_mask, abuf);
    launch_gemm(stream, abuf, vh, nullptr, nullptr, oh,
                T2c, HDc, SEnc, T2c, HDc, HDc, sS, sKc, sQ, ZB, false);  // K=300 -> guarded
    merge_heads<<<gblk((long)M1 * Dm), 256, 0, stream>>>(oh, attnb);
    launch_gemm(stream, attnb, wCO + (long)i * DD, bco + (long)i * Dm, h, h,
                M1, Dm, Dm, Dm, Dm, Dm, 0, 0, 0, 1, false);

    // ================= FFN =================
    ln_kernel<<<M1, 256, 0, stream>>>(h, ln3_s + (long)i * Dm, ln3_b + (long)i * Dm, nullptr, hn);
    launch_gemm(stream, hn, wF1 + (long)i * Dm * DFFc, b1 + (long)i * DFFc, nullptr, tmp,
                M1, DFFc, Dm, Dm, DFFc, DFFc, 0, 0, 0, 1, false);
    gelu_kernel<<<gblk((long)M1 * DFFc), 256, 0, stream>>>(tmp, ge, (long)M1 * DFFc);
    launch_gemm(stream, ge, wF2 + (long)i * Dm * DFFc, b2 + (long)i * Dm, h, h,
                M1, Dm, DFFc, DFFc, Dm, Dm, 0, 0, 0, 1, false);
  }

  // ---- final LN + LM head (first L rows of each batch half; N=50001 -> guarded) ----
  ln_kernel<<<M1, 256, 0, stream>>>(h, lnf_s, lnf_b, nullptr, hn);
  launch_gemm(stream, hn, wLM, nullptr, nullptr, logits,
              Lc, Vp, Dm, Dm, Dm, Vp,
              (long)T2c * Dm, 0, (long)Lc * Vp, Bq, true);

  // ---- weighted masked NLL ----
  zero_acc<<<1, 1, 0, stream>>>(accp);
  loss_kernel<<<Bq * Lc, 256, 0, stream>>>(logits, x0v, tt, mkf, accp);
  final_kernel<<<1, 1, 0, stream>>>(accp, (float*)d_out);
}
